// GCN_NeighSampler_81200651698180
// MI455X (gfx1250) — matrix operand order repaired and checked by reading, not compiled, not run on hardware
//
#include <hip/hip_runtime.h>
#include <hip/hip_bf16.h>
#include <stdint.h>

typedef __attribute__((ext_vector_type(16))) __bf16 v16bf;
typedef __attribute__((ext_vector_type(8)))  float  v8f;

#define K_DIM 128   // inner dim of every GEMM in this net

// ---------------------------------------------------------------------------
// GEMM: C[N x F] = A[N x 128] @ W[128 x F], bf16 WMMA with f32 accumulation.
// F is a compile-time constant (128 or 64) so all addressing constant-folds.
// 8 waves/block; wave w owns column tile w. B fragments live in registers for
// the whole kernel. A tiles (16 rows x 128 f32 = contiguous 8KB) are staged
// into LDS with global_load_async_to_lds_b128 (ASYNCcnt) on full tiles.
// ---------------------------------------------------------------------------
template <int F>
__global__ __launch_bounds__(256) void gemm_wmma_bf16(
    const float* __restrict__ A, const float* __restrict__ W,
    float* __restrict__ C, int N)
{
    __shared__ float ldsAf[16 * K_DIM];           // 8KB raw f32 A tile

    const int lane = threadIdx.x & 31;
    const int wave = threadIdx.x >> 5;
    constexpr int nColTiles = F >> 4;             // 8 (F=128) or 4 (F=64)
    const bool active = wave < nColTiles;

    // ---- per-wave B fragments (W tiny & L2 resident), held in registers ----
    // B k order chosen here (contiguous, same as A): lane n = lane&15 ; lanes>=16 hold K+16 ;
    // VGPR v holds K = 2v, 2v+1 within the 16-K half.
    const int col0  = (active ? wave : 0) * 16;
    const int bn    = col0 + (lane & 15);
    const int bkoff = (lane >> 4) * 16;
    v16bf bfrag[4];
    #pragma unroll
    for (int c = 0; c < 4; ++c) {
        v16bf b;
        const int kc = c * 32 + bkoff;
        #pragma unroll
        for (int v = 0; v < 8; ++v) {
            b[2 * v]     = (__bf16)W[(kc + 2 * v)     * F + bn];
            b[2 * v + 1] = (__bf16)W[(kc + 2 * v + 1) * F + bn];
        }
        bfrag[c] = b;
    }

    const int nRowTiles = (N + 15) >> 4;
    const int m     = lane & 15;
    const int akoff = (lane >> 4) * 16;           // A: high half-wave holds K+16
    const uint32_t ldsBase = (uint32_t)(uintptr_t)&ldsAf[0];

    for (int rt = blockIdx.x; rt < nRowTiles; rt += gridDim.x) {
        const int row0 = rt * 16;
        const bool fullTile = (row0 + 16) <= N;

        __syncthreads();   // WAR: previous iteration's readers done
        if (fullTile) {
            // contiguous 8KB tile: 256 threads x 16B x 2 async copies
            const char*    gsrc = (const char*)(A + (size_t)row0 * K_DIM);
            const uint32_t o    = threadIdx.x * 16u;
            asm volatile("global_load_async_to_lds_b128 %0, %1, off"
                         :: "v"(ldsBase + o),        "v"(gsrc + o)        : "memory");
            asm volatile("global_load_async_to_lds_b128 %0, %1, off"
                         :: "v"(ldsBase + o + 4096), "v"(gsrc + o + 4096) : "memory");
            asm volatile("s_wait_asynccnt 0" ::: "memory");
        } else {
            for (int i = threadIdx.x; i < 16 * K_DIM; i += blockDim.x) {
                const int gr = row0 + (i >> 7);
                ldsAf[i] = (gr < N) ? A[(size_t)gr * K_DIM + (i & 127)] : 0.f;
            }
        }
        __syncthreads();

        if (active) {
            v8f acc = {};
            #pragma unroll
            for (int c = 0; c < 4; ++c) {
                // A k order chosen here (contiguous, same as B): lane m ; VGPR v holds
                // K = 2v, 2v+1 (+16 for lanes>=16) within chunk.
                v16bf a;
                const int kc = c * 32;
                #pragma unroll
                for (int v = 0; v < 8; ++v) {
                    const int k = kc + 2 * v + akoff;
                    const float2 f2 =
                        *reinterpret_cast<const float2*>(&ldsAf[m * K_DIM + k]);
                    a[2 * v]     = (__bf16)f2.x;
                    a[2 * v + 1] = (__bf16)f2.y;
                }
                acc = __builtin_amdgcn_wmma_f32_16x16x32_bf16(
                    false, a, false, bfrag[c], (short)0, acc, false, false);
            }
            // D layout: lane n = lane&15 ; VGPR v = row v (+8 for lanes>=16)
            const int crow = row0 + (lane >> 4) * 8;
            float* cp = C + (size_t)crow * F + col0 + (lane & 15);
            if (fullTile) {
                #pragma unroll
                for (int v = 0; v < 8; ++v) cp[v * F] = acc[v];   // imm offsets
            } else {
                #pragma unroll
                for (int v = 0; v < 8; ++v)
                    if (crow + v < N) cp[v * F] = acc[v];
            }
        }
    }
}

// ---------------------------------------------------------------------------
// agg[n][f] = bias[f]  (scatter-add target initialization, bias folded in)
// ---------------------------------------------------------------------------
__global__ __launch_bounds__(256) void init_bias(
    float* __restrict__ agg, const float* __restrict__ bias, int N, int F)
{
    const size_t total = (size_t)N * F;
    for (size_t i = blockIdx.x * (size_t)blockDim.x + threadIdx.x;
         i < total; i += (size_t)gridDim.x * blockDim.x)
        agg[i] = bias[i & (F - 1)];   // F is a power of two (128 / 64)
}

// ---------------------------------------------------------------------------
// agg[dst[e]] += h[src[e]]  — one wave per edge, f32 atomics (L2 resident),
// next edge's source row prefetched (global_prefetch_b8).
// ---------------------------------------------------------------------------
__global__ __launch_bounds__(256) void scatter_add(
    const float* __restrict__ h, const int* __restrict__ src,
    const int* __restrict__ dst, float* __restrict__ agg, int E, int F)
{
    const int wpb   = blockDim.x >> 5;
    const int wave  = blockIdx.x * wpb + (threadIdx.x >> 5);
    const int lane  = threadIdx.x & 31;
    const int nwave = gridDim.x * wpb;

    for (int e = wave; e < E; e += nwave) {
        const int en = e + nwave;
        if (en < E)
            __builtin_prefetch(h + (size_t)src[en] * F, 0, 1);
        const float* hs = h   + (size_t)src[e] * F;
        float*       ad = agg + (size_t)dst[e] * F;
        for (int f = lane; f < F; f += 32)
            atomicAdd(&ad[f], hs[f]);
    }
}

// ---------------------------------------------------------------------------
// BatchNorm: per-column partial sums / sums-of-squares (blockDim.x == F)
// ---------------------------------------------------------------------------
__global__ void bn_partial(const float* __restrict__ h,
                           float* __restrict__ sums, float* __restrict__ sqs,
                           int N, int F)
{
    const int f = threadIdx.x;
    float s = 0.f, q = 0.f;
    for (int r = blockIdx.x; r < N; r += gridDim.x) {
        const float v = h[(size_t)r * F + f];
        s += v; q += v * v;
    }
    atomicAdd(&sums[f], s);
    atomicAdd(&sqs[f], q);
}

__global__ void bn_finalize(const float* __restrict__ sums,
                            const float* __restrict__ sqs,
                            const float* __restrict__ g,
                            const float* __restrict__ be,
                            float* __restrict__ scale, float* __restrict__ shift,
                            int N, int F)
{
    const int f = blockIdx.x * blockDim.x + threadIdx.x;
    if (f < F) {
        const float m  = sums[f] / (float)N;
        const float v  = sqs[f] / (float)N - m * m;
        const float sc = g[f] * rsqrtf(v + 1e-5f);
        scale[f] = sc;
        shift[f] = be[f] - m * sc;
    }
}

__global__ __launch_bounds__(256) void bn_apply_relu(
    float* __restrict__ h, const float* __restrict__ scale,
    const float* __restrict__ shift, int N, int F)
{
    const size_t total = (size_t)N * F;
    for (size_t i = blockIdx.x * (size_t)blockDim.x + threadIdx.x;
         i < total; i += (size_t)gridDim.x * blockDim.x) {
        const int f = (int)(i & (F - 1));
        h[i] = fmaxf(0.f, h[i] * scale[f] + shift[f]);
    }
}

__global__ void zero_f32(float* __restrict__ p, int n)
{
    const int i = blockIdx.x * blockDim.x + threadIdx.x;
    if (i < n) p[i] = 0.f;
}

// ---------------------------------------------------------------------------
// log_softmax over rows of [R x 64]: one wave per row, 2 elems per lane
// ---------------------------------------------------------------------------
__global__ __launch_bounds__(256) void log_softmax64(
    const float* __restrict__ in, float* __restrict__ out, int R)
{
    const int wpb   = blockDim.x >> 5;
    const int wave  = blockIdx.x * wpb + (threadIdx.x >> 5);
    const int lane  = threadIdx.x & 31;
    const int nwave = gridDim.x * wpb;

    for (int r = wave; r < R; r += nwave) {
        const float* row = in + (size_t)r * 64;
        const float v0 = row[lane], v1 = row[lane + 32];
        float mx = fmaxf(v0, v1);
        #pragma unroll
        for (int o = 16; o > 0; o >>= 1) mx = fmaxf(mx, __shfl_xor(mx, o, 32));
        float e = __expf(v0 - mx) + __expf(v1 - mx);
        #pragma unroll
        for (int o = 16; o > 0; o >>= 1) e += __shfl_xor(e, o, 32);
        const float lse = __logf(e) + mx;
        out[(size_t)r * 64 + lane]      = v0 - lse;
        out[(size_t)r * 64 + lane + 32] = v1 - lse;
    }
}

// ---------------------------------------------------------------------------
extern "C" void kernel_launch(void* const* d_in, const int* in_sizes, int n_in,
                              void* d_out, int out_size, void* d_ws, size_t ws_size,
                              hipStream_t stream)
{
    const float* x    = (const float*)d_in[0];
    const int*   ei   = (const int*)  d_in[1];
    const float* W1   = (const float*)d_in[2];
    const float* b1   = (const float*)d_in[3];
    const float* W2   = (const float*)d_in[4];
    const float* b2   = (const float*)d_in[5];
    const float* W3   = (const float*)d_in[6];
    const float* b3   = (const float*)d_in[7];
    const float* g1   = (const float*)d_in[8];
    const float* be1  = (const float*)d_in[9];
    const float* g2   = (const float*)d_in[10];
    const float* be2  = (const float*)d_in[11];
    float* out = (float*)d_out;

    const int N   = in_sizes[0] / 128;
    const int E   = in_sizes[1] / 2;
    const int TGT = out_size / 64;
    const int* src = ei;
    const int* dst = ei + E;

    // workspace layout
    float* bufA  = (float*)d_ws;                  // N*128
    float* bufB  = bufA + (size_t)N * 128;        // N*128
    float* sums  = bufB + (size_t)N * 128;        // 128
    float* sqs   = sums + 128;                    // 128
    float* scale = sqs  + 128;                    // 128
    float* shift = scale + 128;                   // 128
    float* agg3  = bufA + (size_t)N * 64;         // N*64 (upper half of bufA)

    const int nRowTiles = (N + 15) / 16;
    const dim3 B256(256);
    const dim3 GEMM_G(nRowTiles < 2048 ? nRowTiles : 2048);
    const dim3 EW_G(4096);        // grid-stride elementwise
    const dim3 SC_G(4096);        // scatter: 8 waves/block
    const dim3 BN_G(1024);

    // ---- layer 1: t1 = x@W1 ; agg1 = scatter(t1)+b1 ; BN+ReLU -> bufB ----
    gemm_wmma_bf16<128><<<GEMM_G, B256, 0, stream>>>(x, W1, bufA, N);
    init_bias<<<EW_G, B256, 0, stream>>>(bufB, b1, N, 128);
    scatter_add<<<SC_G, B256, 0, stream>>>(bufA, src, dst, bufB, E, 128);
    zero_f32<<<1, 256, 0, stream>>>(sums, 256);
    bn_partial<<<BN_G, dim3(128), 0, stream>>>(bufB, sums, sqs, N, 128);
    bn_finalize<<<1, 128, 0, stream>>>(sums, sqs, g1, be1, scale, shift, N, 128);
    bn_apply_relu<<<EW_G, B256, 0, stream>>>(bufB, scale, shift, N, 128);

    // ---- layer 2: t2 = h1@W2 ; agg2 = scatter(t2)+b2 ; BN+ReLU -> bufB ----
    gemm_wmma_bf16<128><<<GEMM_G, B256, 0, stream>>>(bufB, W2, bufA, N);
    init_bias<<<EW_G, B256, 0, stream>>>(bufB, b2, N, 128);
    scatter_add<<<SC_G, B256, 0, stream>>>(bufA, src, dst, bufB, E, 128);
    zero_f32<<<1, 256, 0, stream>>>(sums, 256);
    bn_partial<<<BN_G, dim3(128), 0, stream>>>(bufB, sums, sqs, N, 128);
    bn_finalize<<<1, 128, 0, stream>>>(sums, sqs, g2, be2, scale, shift, N, 128);
    bn_apply_relu<<<EW_G, B256, 0, stream>>>(bufB, scale, shift, N, 128);

    // ---- layer 3: t3 = h2@W3 (N x 64) ; agg3 = scatter(t3)+b3 ----
    gemm_wmma_bf16<64><<<GEMM_G, B256, 0, stream>>>(bufB, W3, bufA, N);
    init_bias<<<EW_G, B256, 0, stream>>>(agg3, b3, N, 64);
    scatter_add<<<SC_G, B256, 0, stream>>>(bufA, src, dst, agg3, E, 64);

    // ---- log_softmax over first TGT rows ----
    log_softmax64<<<dim3(2048), B256, 0, stream>>>(agg3, out, TGT);
}
